// GraphSSSM_53360673686109
// MI455X (gfx1250) — matrix operand order repaired and checked by reading, not compiled, not run on hardware
//
#include <hip/hip_runtime.h>
#include <hip/hip_bf16.h>

typedef __bf16 bf16;
typedef __attribute__((ext_vector_type(16))) __bf16 v16bf;
typedef __attribute__((ext_vector_type(8)))  float  v8f;

#define BATCH  8
#define SEQ    1024
#define DMODEL 256
#define DINNER 256
#define DSTATE 16
#define MFLAT  (DINNER*DSTATE)   // 4096
#define PROJ2  (2*DINNER)        // 512
#define SPJ    (1+2*DSTATE)      // 33
#define ROWS   (BATCH*SEQ)       // 8192

union V16U { uint4 q[2]; v16bf v; };
union V8U  { uint4 q;    bf16  h[8]; };
union V4U  { uint2 q;    bf16  h[4]; };

// ---- packed fragment loads (2x ds_load_b128 each) ----
// A fragment: LDS tile stored m-major As[m*ld + k]; element e -> k = 16*(e/8) + 8*half + e%8
static __device__ __forceinline__ v16bf frag_a(const bf16* As, int ld, int m, int half) {
    V16U u;
    const bf16* p = As + m * ld + half * 8;
    u.q[0] = *(const uint4*)(p);
    u.q[1] = *(const uint4*)(p + 16);
    return u.v;
}
// B fragment: LDS tile stored n-major Bs[n*ld + k]; element e -> k = 16*(e/8) + 8*half + e%8
static __device__ __forceinline__ v16bf frag_b(const bf16* Bs, int ld, int n, int half) {
    V16U u;
    const bf16* p = Bs + n * ld + half * 8;
    u.q[0] = *(const uint4*)(p);
    u.q[1] = *(const uint4*)(p + 16);
    return u.v;
}
static __device__ __forceinline__ v8f wmma_bf16(v16bf a, v16bf b, v8f c) {
    return __builtin_amdgcn_wmma_f32_16x16x32_bf16(false, a, false, b, (short)0, c, false, false);
}

// ---- CDNA5 async global->LDS copy (ASYNCcnt) ----
static __device__ __forceinline__ void async_copy_b128(void* lds_dst, const void* gsrc) {
    unsigned lds_off = (unsigned)(unsigned long long)lds_dst;  // LDS aperture: low 32 bits
    asm volatile("global_load_async_to_lds_b128 %0, %1, off"
                 :: "v"(lds_off), "v"(gsrc) : "memory");
}
static __device__ __forceinline__ void wait_async_all() {
    asm volatile("s_wait_asynccnt 0x0" ::: "memory");
}

// ---------- Kernel 0: E (fp32) -> E_bf [n][k] and Et_bf [k][n] (bf16) ----------
__global__ __launch_bounds__(256)
void k_ecvt(const float* __restrict__ eig, bf16* __restrict__ ebf, bf16* __restrict__ ebt) {
    __shared__ bf16 Ts[32 * 36];
    const int b = blockIdx.z, n0 = blockIdx.x * 32, k0 = blockIdx.y * 32;
    const int tid = threadIdx.x;
    const float* E = eig + (size_t)b * SEQ * SEQ;
    {
        int r = tid >> 3, c4 = (tid & 7) * 4;
        float4 f = *(const float4*)&E[(size_t)(n0 + r) * SEQ + k0 + c4];
        V4U p;
        p.h[0] = (bf16)f.x; p.h[1] = (bf16)f.y; p.h[2] = (bf16)f.z; p.h[3] = (bf16)f.w;
        *(uint2*)&ebf[(size_t)b * SEQ * SEQ + (size_t)(n0 + r) * SEQ + k0 + c4] = p.q;
        *(uint2*)&Ts[r * 36 + c4] = p.q;
    }
    __syncthreads();
    {
        int rk = tid >> 3, cn = (tid & 7) * 4;
        V4U o;
#pragma unroll
        for (int j = 0; j < 4; ++j) o.h[j] = Ts[(cn + j) * 36 + rk];
        *(uint2*)&ebt[(size_t)b * SEQ * SEQ + (size_t)(k0 + rk) * SEQ + n0 + cn] = o.q;
    }
}

// ---------- Kernel 0b: weight [K][N] fp32 -> [N][K] bf16 ----------
__global__ __launch_bounds__(256)
void k_wtr(const float* __restrict__ w, bf16* __restrict__ wt, int K, int N) {
    __shared__ bf16 T[32 * 36];
    const int k0 = blockIdx.x * 32, n0 = blockIdx.y * 32;
    const int tid = threadIdx.x;
    {
        int r = tid >> 3, c4 = (tid & 7) * 4;
        float4 f = *(const float4*)&w[(size_t)(k0 + r) * N + n0 + c4];
        V4U p;
        p.h[0] = (bf16)f.x; p.h[1] = (bf16)f.y; p.h[2] = (bf16)f.z; p.h[3] = (bf16)f.w;
        *(uint2*)&T[r * 36 + c4] = p.q;
    }
    __syncthreads();
    {
        int rn = tid >> 3, ck = (tid & 7) * 4;
        V4U o;
#pragma unroll
        for (int j = 0; j < 4; ++j) o.h[j] = T[(ck + j) * 36 + rn];
        *(uint2*)&wt[(size_t)(n0 + rn) * K + k0 + ck] = o.q;
    }
}

// ---------- Kernel 1: xz = x @ in_proj_w + b  (8192x256 @ 256x512), tile 128x64 ----------
__global__ __launch_bounds__(256)
void k_in_proj(const float* __restrict__ x, const bf16* __restrict__ wt,
               const float* __restrict__ bias, float* __restrict__ xz) {
    __shared__ bf16 As[128 * 40]; // [m][k]
    __shared__ bf16 Bs[64 * 40];  // [n][k]
    const int tid = threadIdx.x, lane = tid & 31, wv = tid >> 5;
    const int lr = lane & 15, half = lane >> 4;
    const int row0 = blockIdx.x * 128, col0 = blockIdx.y * 64;
    v8f c[4] = {};
    for (int k0 = 0; k0 < DMODEL; k0 += 32) {
#pragma unroll
        for (int it = 0; it < 2; ++it) {           // A: 128x32 f32 -> bf16, vectorized
            int g = tid + it * 256;
            int ml = g >> 2, kq = (g & 3) * 8;
            const float4* f = (const float4*)&x[(size_t)(row0 + ml) * DMODEL + k0 + kq];
            float4 f0 = f[0], f1 = f[1];
            V8U u;
            u.h[0] = (bf16)f0.x; u.h[1] = (bf16)f0.y; u.h[2] = (bf16)f0.z; u.h[3] = (bf16)f0.w;
            u.h[4] = (bf16)f1.x; u.h[5] = (bf16)f1.y; u.h[6] = (bf16)f1.z; u.h[7] = (bf16)f1.w;
            *(uint4*)&As[ml * 40 + kq] = u.q;
        }
        {   // B: pure uint4 copy from pre-transposed bf16 weights
            int nl = tid >> 2, kq = (tid & 3) * 8;
            *(uint4*)&Bs[nl * 40 + kq] = *(const uint4*)&wt[(size_t)(col0 + nl) * DMODEL + k0 + kq];
        }
        __syncthreads();
        v16bf a = frag_a(As, 40, wv * 16 + lr, half);
        v16bf bb[4];
#pragma unroll
        for (int j = 0; j < 4; ++j) bb[j] = frag_b(Bs, 40, j * 16 + lr, half);
#pragma unroll
        for (int j = 0; j < 4; ++j) c[j] = wmma_bf16(a, bb[j], c[j]);
        __syncthreads();
    }
#pragma unroll
    for (int j = 0; j < 4; ++j)
#pragma unroll
        for (int v = 0; v < 8; ++v) {
            int r = row0 + wv * 16 + v + 8 * half;
            int cc = col0 + j * 16 + lr;
            xz[(size_t)r * PROJ2 + cc] = c[j][v] + bias[cc];
        }
}

// ---------- Kernel 2: gate + s_proj + softplus + XBf(bf16) + Csel ----------
__global__ __launch_bounds__(256)
void k_gate_sproj(const float* __restrict__ xz, const float* __restrict__ spw,
                  const float* __restrict__ spb, const float* __restrict__ pdelta,
                  float* __restrict__ xs, float* __restrict__ csel,
                  bf16* __restrict__ xbf) {
    __shared__ float sxs[DINNER];
    __shared__ float sdbc[SPJ];
    const int row = blockIdx.x;          // b*SEQ + t
    const int d = threadIdx.x;
    const float x1 = xz[(size_t)row * PROJ2 + d];
    const float z  = xz[(size_t)row * PROJ2 + DINNER + d];
    const float g  = x1 * (z / (1.f + __expf(-z)));   // x1 * silu(z)
    sxs[d] = g;
    xs[(size_t)row * DINNER + d] = g;
    __syncthreads();
    if (d < SPJ) {
        float acc = spb[d];
        for (int k = 0; k < DINNER; ++k) acc += sxs[k] * spw[k * SPJ + d];
        sdbc[d] = acc;
    }
    __syncthreads();
    if (d < DSTATE) csel[(size_t)row * DSTATE + d] = sdbc[1 + DSTATE + d];
    const float draw  = pdelta[d] + sdbc[0];
    const float delta = (draw > 20.f) ? draw : log1pf(__expf(draw));  // softplus
    const float gd = g * delta;
    V8U lo, hi;
#pragma unroll
    for (int s = 0; s < 8; ++s) {
        lo.h[s] = (bf16)(gd * sdbc[1 + s]);
        hi.h[s] = (bf16)(gd * sdbc[9 + s]);
    }
    uint4* dst = (uint4*)(xbf + (size_t)row * MFLAT + d * DSTATE);
    dst[0] = lo.q;
    dst[1] = hi.q;
}

// ---------- Kernel 2b: XBf [n][m] -> XBfT [m][n] (bf16 tiled transpose) ----------
__global__ __launch_bounds__(256)
void k_xtr(const bf16* __restrict__ xbf, bf16* __restrict__ xbft) {
    __shared__ bf16 T[64 * 80];
    const int b = blockIdx.z, n0 = blockIdx.x * 64, m0 = blockIdx.y * 64;
    const int tid = threadIdx.x;
    const bf16* X  = xbf  + (size_t)b * SEQ * MFLAT;
    bf16*       XT = xbft + (size_t)b * MFLAT * SEQ;
#pragma unroll
    for (int it = 0; it < 2; ++it) {
        int g = tid + it * 256;
        int nl = g >> 3, mq = (g & 7) * 8;
        *(uint4*)&T[nl * 80 + mq] = *(const uint4*)&X[(size_t)(n0 + nl) * MFLAT + m0 + mq];
    }
    __syncthreads();
#pragma unroll
    for (int it = 0; it < 2; ++it) {
        int g = tid + it * 256;
        int ml = g >> 3, nq = (g & 7) * 8;
        V8U o;
#pragma unroll
        for (int j = 0; j < 8; ++j) o.h[j] = T[(nq + j) * 80 + ml];
        *(uint4*)&XT[(size_t)(m0 + ml) * SEQ + n0 + nq] = o.q;
    }
}

// ---- Kernel 3: QTXT[m,kk] = (1/EigVals[kk]) * sum_n Et[kk,n]*XBfT[m,n] ----
// tile 128kk x 64m, async double-buffered staging
__global__ __launch_bounds__(256)
void k_qtx(const bf16* __restrict__ ebt, const float* __restrict__ eigvals,
           const bf16* __restrict__ xbft, bf16* __restrict__ qtxt) {
    __shared__ bf16 As[2][128 * 40];  // [kk][n]
    __shared__ bf16 Bs[2][64 * 40];   // [m][n]
    __shared__ bf16 Ts[64 * 136];     // output transpose staging [m][kk]
    const int b = blockIdx.z;
    const int kk0 = blockIdx.x * 128, m0 = blockIdx.y * 64;
    const int tid = threadIdx.x, lane = tid & 31, wv = tid >> 5;
    const int lr = lane & 15, half = lane >> 4;
    const bf16* ET = ebt  + (size_t)b * SEQ * SEQ;
    const bf16* XT = xbft + (size_t)b * MFLAT * SEQ;
    const int aml = tid >> 1, aseg = (tid & 1) * 16;   // A staging: 2 asyncs/thread
    const int bml = tid >> 2, bnq  = (tid & 3) * 8;    // B staging: 1 async/thread
    auto issue = [&](int buf, int n0) {
        const bf16* sa = &ET[(size_t)(kk0 + aml) * SEQ + n0 + aseg];
        async_copy_b128(&As[buf][aml * 40 + aseg],     sa);
        async_copy_b128(&As[buf][aml * 40 + aseg + 8], sa + 8);
        async_copy_b128(&Bs[buf][bml * 40 + bnq], &XT[(size_t)(m0 + bml) * SEQ + n0 + bnq]);
    };
    v8f c[4] = {};
    issue(0, 0);
    for (int s = 0; s < SEQ / 32; ++s) {
        wait_async_all();
        __syncthreads();
        if (s + 1 < SEQ / 32) issue((s + 1) & 1, (s + 1) * 32);
        const bf16* Ab = As[s & 1];
        const bf16* Bb = Bs[s & 1];
        v16bf a = frag_a(Ab, 40, wv * 16 + lr, half);
        v16bf bb[4];
#pragma unroll
        for (int j = 0; j < 4; ++j) bb[j] = frag_b(Bb, 40, j * 16 + lr, half);
#pragma unroll
        for (int j = 0; j < 4; ++j) c[j] = wmma_bf16(a, bb[j], c[j]);
    }
    // epilogue: scale rows by 1/EigVals, pack 8 consecutive kk per lane, transpose via LDS
    float inv[8];
#pragma unroll
    for (int v = 0; v < 8; ++v)
        inv[v] = 1.f / eigvals[b * SEQ + kk0 + wv * 16 + 8 * half + v];
#pragma unroll
    for (int j = 0; j < 4; ++j) {
        V8U u;
#pragma unroll
        for (int v = 0; v < 8; ++v) u.h[v] = (bf16)(c[j][v] * inv[v]);
        *(uint4*)&Ts[(j * 16 + lr) * 136 + wv * 16 + 8 * half] = u.q;
    }
    __syncthreads();
#pragma unroll
    for (int it = 0; it < 4; ++it) {       // 64m x 128kk -> global, coalesced uint4
        int g = tid + it * 256;
        int mr = g >> 4, c8 = (g & 15) * 8;
        *(uint4*)&qtxt[((size_t)b * MFLAT + m0 + mr) * SEQ + kk0 + c8] =
            *(const uint4*)&Ts[mr * 136 + c8];
    }
}

// ---- Kernel 4: MX = E @ QTX (via QTXT), fused Csel contraction + silu -> ybuf ----
// tile 64t x 64m, async double-buffered staging
__global__ __launch_bounds__(256)
void k_mx(const bf16* __restrict__ ebf, const bf16* __restrict__ qtxt,
          const float* __restrict__ csel, const float* __restrict__ xs,
          bf16* __restrict__ ybuf) {
    __shared__ bf16 As[2][64 * 40];   // [t][k]
    __shared__ bf16 Bs[2][64 * 40];   // [m][k]
    __shared__ float red[8][16][17];
    const int b = blockIdx.z;
    const int t0 = blockIdx.x * 64, m0 = blockIdx.y * 64;
    const int tid = threadIdx.x, lane = tid & 31, wv = tid >> 5;
    const int lr = lane & 15, half = lane >> 4;
    const int tr = wv >> 1, cg = wv & 1;
    const bf16* E = ebf  + (size_t)b * SEQ * SEQ;
    const bf16* Q = qtxt + (size_t)b * MFLAT * SEQ;
    const int stl = tid >> 2, skq = (tid & 3) * 8;     // 1 async/thread per tile
    auto issue = [&](int buf, int k0) {
        async_copy_b128(&As[buf][stl * 40 + skq], &E[(size_t)(t0 + stl) * SEQ + k0 + skq]);
        async_copy_b128(&Bs[buf][stl * 40 + skq], &Q[(size_t)(m0 + stl) * SEQ + k0 + skq]);
    };
    v8f c[2] = {};
    issue(0, 0);
    for (int s = 0; s < SEQ / 32; ++s) {
        wait_async_all();
        __syncthreads();
        if (s + 1 < SEQ / 32) issue((s + 1) & 1, (s + 1) * 32);
        const bf16* Ab = As[s & 1];
        const bf16* Bb = Bs[s & 1];
        v16bf a = frag_a(Ab, 40, tr * 16 + lr, half);
        v16bf bb[2];
#pragma unroll
        for (int cc = 0; cc < 2; ++cc) bb[cc] = frag_b(Bb, 40, (cg * 2 + cc) * 16 + lr, half);
#pragma unroll
        for (int cc = 0; cc < 2; ++cc) c[cc] = wmma_bf16(a, bb[cc], c[cc]);
    }
#pragma unroll
    for (int cc = 0; cc < 2; ++cc) {
        __syncthreads();
#pragma unroll
        for (int v = 0; v < 8; ++v) {
            int ti = v + 8 * half;
            int t  = t0 + tr * 16 + ti;
            red[wv][ti][lr] = c[cc][v] * csel[((size_t)b * SEQ + t) * DSTATE + lr];
        }
        __syncthreads();
        if (lane < 16) {
            int ti = lane;
            float s = 0.f;
#pragma unroll
            for (int j = 0; j < 16; ++j) s += red[wv][ti][j];
            int t = t0 + tr * 16 + ti;
            int d = blockIdx.y * 4 + cg * 2 + cc;
            float u = xs[((size_t)b * SEQ + t) * DINNER + d] + s;   // xs + MXC
            float y = u / (1.f + __expf(-u));                       // silu
            ybuf[((size_t)b * SEQ + t) * DINNER + d] = (bf16)y;
        }
    }
}

// ---------- Kernel 5: out = ybuf @ out_proj_w + b  (8192x256 @ 256x256), tile 128x64 ----------
__global__ __launch_bounds__(256)
void k_out_proj(const bf16* __restrict__ ybuf, const bf16* __restrict__ wt,
                const float* __restrict__ bias, float* __restrict__ out) {
    __shared__ bf16 As[128 * 40];
    __shared__ bf16 Bs[64 * 40];
    const int tid = threadIdx.x, lane = tid & 31, wv = tid >> 5;
    const int lr = lane & 15, half = lane >> 4;
    const int row0 = blockIdx.x * 128, col0 = blockIdx.y * 64;
    v8f c[4] = {};
    for (int k0 = 0; k0 < DINNER; k0 += 32) {
#pragma unroll
        for (int it = 0; it < 2; ++it) {   // A: pure uint4 copy (128x32)
            int g = tid + it * 256;
            int ml = g >> 2, kq = (g & 3) * 8;
            *(uint4*)&As[ml * 40 + kq] =
                *(const uint4*)&ybuf[(size_t)(row0 + ml) * DINNER + k0 + kq];
        }
        {   // B: pure uint4 copy from pre-transposed bf16 weights
            int nl = tid >> 2, kq = (tid & 3) * 8;
            *(uint4*)&Bs[nl * 40 + kq] = *(const uint4*)&wt[(size_t)(col0 + nl) * DINNER + k0 + kq];
        }
        __syncthreads();
        v16bf a = frag_a(As, 40, wv * 16 + lr, half);
        v16bf bb[4];
#pragma unroll
        for (int j = 0; j < 4; ++j) bb[j] = frag_b(Bs, 40, j * 16 + lr, half);
#pragma unroll
        for (int j = 0; j < 4; ++j) c[j] = wmma_bf16(a, bb[j], c[j]);
        __syncthreads();
    }
#pragma unroll
    for (int j = 0; j < 4; ++j)
#pragma unroll
        for (int v = 0; v < 8; ++v) {
            int r = row0 + wv * 16 + v + 8 * half;
            int cc = col0 + j * 16 + lr;
            out[(size_t)r * DMODEL + cc] = c[j][v] + bias[cc];
        }
}

extern "C" void kernel_launch(void* const* d_in, const int* in_sizes, int n_in,
                              void* d_out, int out_size, void* d_ws, size_t ws_size,
                              hipStream_t stream) {
    const float* x       = (const float*)d_in[0];
    const float* eig     = (const float*)d_in[2];
    const float* eigvals = (const float*)d_in[3];
    const float* w_in    = (const float*)d_in[4];
    const float* b_in    = (const float*)d_in[5];
    const float* w_sp    = (const float*)d_in[6];
    const float* b_sp    = (const float*)d_in[7];
    const float* pdelta  = (const float*)d_in[8];
    const float* w_out   = (const float*)d_in[10];
    const float* b_out   = (const float*)d_in[11];
    float* out = (float*)d_out;

    char* ws = (char*)d_ws;
    size_t off = 0;
    float* xz   = (float*)(ws + off); off += (size_t)ROWS * PROJ2  * sizeof(float); // 16 MB
    float* xs   = (float*)(ws + off); off += (size_t)ROWS * DINNER * sizeof(float); //  8 MB
    float* csel = (float*)(ws + off); off += (size_t)ROWS * DSTATE * sizeof(float); // .5 MB
    bf16*  xbf  = (bf16*)(ws + off);  off += (size_t)ROWS * MFLAT  * sizeof(bf16);  // 64 MB
    bf16*  xbft = (bf16*)(ws + off);  off += (size_t)ROWS * MFLAT  * sizeof(bf16);  // 64 MB
    bf16*  qtxt = (bf16*)(ws + off);  off += (size_t)ROWS * MFLAT  * sizeof(bf16);  // 64 MB
    bf16*  ybuf = (bf16*)(ws + off);  off += (size_t)ROWS * DINNER * sizeof(bf16);  //  4 MB
    bf16*  ebf  = (bf16*)(ws + off);  off += (size_t)BATCH * SEQ * SEQ * sizeof(bf16); // 16.8 MB
    bf16*  ebt  = (bf16*)(ws + off);  off += (size_t)BATCH * SEQ * SEQ * sizeof(bf16); // 16.8 MB
    bf16*  wtin = (bf16*)(ws + off);  off += (size_t)DMODEL * PROJ2 * sizeof(bf16);    // 256 KB
    bf16*  wtout= (bf16*)(ws + off);  off += (size_t)DINNER * DMODEL * sizeof(bf16);   // 128 KB
    (void)ws_size; (void)in_sizes; (void)n_in; (void)out_size;

    k_ecvt      <<<dim3(SEQ / 32, SEQ / 32, BATCH), 256, 0, stream>>>(eig, ebf, ebt);
    k_wtr       <<<dim3(DMODEL / 32, PROJ2 / 32), 256, 0, stream>>>(w_in, wtin, DMODEL, PROJ2);
    k_wtr       <<<dim3(DINNER / 32, DMODEL / 32), 256, 0, stream>>>(w_out, wtout, DINNER, DMODEL);
    k_in_proj   <<<dim3(ROWS / 128, PROJ2 / 64), 256, 0, stream>>>(x, wtin, b_in, xz);
    k_gate_sproj<<<dim3(ROWS), 256, 0, stream>>>(xz, w_sp, b_sp, pdelta, xs, csel, xbf);
    k_xtr       <<<dim3(SEQ / 64, MFLAT / 64, BATCH), 256, 0, stream>>>(xbf, xbft);
    k_qtx       <<<dim3(SEQ / 128, MFLAT / 64, BATCH), 256, 0, stream>>>(ebt, eigvals, xbft, qtxt);
    k_mx        <<<dim3(SEQ / 64, MFLAT / 64, BATCH), 256, 0, stream>>>(ebf, qtxt, csel, xs, ybuf);
    k_out_proj  <<<dim3(ROWS / 128, DMODEL / 64), 256, 0, stream>>>(ybuf, wtout, b_out, out);
}
